// qLinear_738734375753
// MI455X (gfx1250) — matrix operand order repaired and checked by reading, not compiled, not run on hardware
//
#include <hip/hip_runtime.h>

#define IN_F  128
#define OUT_F 128

typedef __attribute__((ext_vector_type(16))) __bf16          v16bf;
typedef __attribute__((ext_vector_type(16))) unsigned short  v16us;
typedef __attribute__((ext_vector_type(8)))  float           v8f;
typedef __attribute__((ext_vector_type(4)))  float           v4f;

__device__ __forceinline__ unsigned short bf16_rne(float f) {
    unsigned int u = __float_as_uint(f);
    u += 0x7fffu + ((u >> 16) & 1u);        // round-to-nearest-even to bf16
    return (unsigned short)(u >> 16);
}
__device__ __forceinline__ float bf16_to_f32(unsigned short h) {
    return __uint_as_float(((unsigned int)h) << 16);
}

// Split one float into (hi = truncated bf16, lo = rne bf16 of residual).
// Truncating hi is lossless overall: the dropped bits land in the residual,
// which lo captures; combined error stays ~2^-17 relative.
__device__ __forceinline__ void split2(float f, unsigned short& hi, unsigned short& lo) {
    unsigned int u = __float_as_uint(f);
    hi = (unsigned short)(u >> 16);
    float r = f - __uint_as_float(u & 0xffff0000u);
    lo = bf16_rne(r);
}

// Build one 16x32 bf16 A-fragment pair (hi/lo split) from an fp32 row.
// Layout: lanes 0-15 K base, lanes 16-31 K+16 (koff handles that);
// elements 0..15 -> K = koff..koff+15 (contiguous, matches B).
__device__ __forceinline__ void load_split_a(const float* __restrict__ xrow, int koff,
                                             v16bf& a_hi, v16bf& a_lo) {
    v4f f0 = *(const v4f*)(xrow + koff);
    v4f f1 = *(const v4f*)(xrow + koff + 4);
    v4f f2 = *(const v4f*)(xrow + koff + 8);
    v4f f3 = *(const v4f*)(xrow + koff + 12);
    v16us hi, lo;
    #pragma unroll
    for (int j = 0; j < 4; ++j) {
        unsigned short h, l;
        split2(f0[j], h, l); hi[j]      = h; lo[j]      = l;
        split2(f1[j], h, l); hi[j + 4]  = h; lo[j + 4]  = l;
        split2(f2[j], h, l); hi[j + 8]  = h; lo[j + 8]  = l;
        split2(f3[j], h, l); hi[j + 12] = h; lo[j + 12] = l;
    }
    a_hi = __builtin_bit_cast(v16bf, hi);
    a_lo = __builtin_bit_cast(v16bf, lo);
}

__global__ __launch_bounds__(256) void qlinear_bf16x3_kernel(
        const float* __restrict__ x, const float* __restrict__ W,
        float* __restrict__ out, int n_rows)
{
    // W split into hi/lo bf16 planes in LDS (64 KB; WGP has 320 KB).
    __shared__ unsigned short w_hi[OUT_F * IN_F];
    __shared__ unsigned short w_lo[OUT_F * IN_F];

    const int tid = threadIdx.x;
    for (int i = tid * 4; i < OUT_F * IN_F; i += 256 * 4) {
        v4f w = *(const v4f*)(W + i);
        #pragma unroll
        for (int j = 0; j < 4; ++j) {
            unsigned short h, l;
            split2(w[j], h, l);
            w_hi[i + j] = h;
            w_lo[i + j] = l;
        }
    }
    __syncthreads();

    const int lane = tid & 31;
    const int wave = tid >> 5;
    const int tile = blockIdx.x * 8 + wave;   // 16 rows of x per wave
    const int row0 = tile * 16;
    if (row0 >= n_rows) return;               // wave-uniform exit (EXEC all-1 for WMMA)

    const int m    = lane & 15;               // A row-in-tile / B,C column-in-tile
    const int half = lane >> 4;

    int arow = row0 + m;                      // clamp tail rows so all lanes load
    if (arow > n_rows - 1) arow = n_rows - 1;
    const float* xrow = x + (long)arow * IN_F;

    v8f acc[8] = {};                          // 8 column tiles of 16x16 f32

    #pragma unroll
    for (int c = 0; c < 4; ++c) {             // K in chunks of 32
        const int koff = c * 32 + half * 16;
        v16bf a_hi, a_lo;
        load_split_a(xrow, koff, a_hi, a_lo);

        // B fragment (32x16 bf16) from LDS: lanes 0-15 K..K+15,
        // lanes 16-31 K+16..K+31, column n = lane%16 (W row-major = Wᵀ cols)
        const int kb = c * 32 + half * 16;
        #pragma unroll
        for (int t = 0; t < 8; ++t) {
            const int nrow = t * 16 + m;
            v16bf b_hi = __builtin_bit_cast(v16bf, *(const v16us*)(&w_hi[nrow * IN_F + kb]));
            v16bf b_lo = __builtin_bit_cast(v16bf, *(const v16us*)(&w_lo[nrow * IN_F + kb]));
            // bf16x3: hi*lo + lo*hi + hi*hi (small terms first), fp32 accumulate
            acc[t] = __builtin_amdgcn_wmma_f32_16x16x32_bf16(
                         false, a_hi, false, b_lo, (short)0, acc[t], false, false);
            acc[t] = __builtin_amdgcn_wmma_f32_16x16x32_bf16(
                         false, a_lo, false, b_hi, (short)0, acc[t], false, false);
            acc[t] = __builtin_amdgcn_wmma_f32_16x16x32_bf16(
                         false, a_hi, false, b_hi, (short)0, acc[t], false, false);
        }
    }

    // C layout: VGPR r -> row (r + half*8), col = t*16 + m
    if (row0 + 16 <= n_rows) {
        #pragma unroll
        for (int t = 0; t < 8; ++t) {
            #pragma unroll
            for (int r = 0; r < 8; ++r) {
                const long row = row0 + r + half * 8;
                out[row * OUT_F + t * 16 + m] = acc[t][r];
            }
        }
    } else {
        #pragma unroll
        for (int t = 0; t < 8; ++t) {
            #pragma unroll
            for (int r = 0; r < 8; ++r) {
                const int row = row0 + r + half * 8;
                if (row < n_rows)
                    out[(long)row * OUT_F + t * 16 + m] = acc[t][r];
            }
        }
    }
}

extern "C" void kernel_launch(void* const* d_in, const int* in_sizes, int n_in,
                              void* d_out, int out_size, void* d_ws, size_t ws_size,
                              hipStream_t stream) {
    const float* x = (const float*)d_in[0];
    const float* W = (const float*)d_in[1];
    float* out = (float*)d_out;
    const int n_rows = in_sizes[0] / IN_F;          // 1,000,000
    const int rows_per_block = 8 * 16;              // 8 waves x 16 rows
    const int blocks = (n_rows + rows_per_block - 1) / rows_per_block;
    qlinear_bf16x3_kernel<<<blocks, 256, 0, stream>>>(x, W, out, n_rows);
}
